// CustomTransformerEncoderLayer_53455162966541
// MI455X (gfx1250) — matrix operand order repaired and checked by reading, not compiled, not run on hardware
//
#include <hip/hip_runtime.h>
#include <hip/hip_bf16.h>

// ---------------------------------------------------------------------------
// CustomTransformerEncoderLayer on MI455X (gfx1250, wave32, WMMA).
// Fixed shapes: B=2, T=2048, D=1024, H=16, DH=64, F=4096.
// Matrix math via v_wmma_f32_16x16x32_f16 (fp32 accum); GEMM staging via
// async global->LDS (ASYNCcnt) with double-buffered LDS.
// ---------------------------------------------------------------------------

typedef __attribute__((ext_vector_type(16))) _Float16 v16h;
typedef __attribute__((ext_vector_type(8)))  float    v8f;

#define TDIM 2048
#define BDIM 2
#define DDIM 1024
#define HDIM 16
#define DHD  64
#define FDIM 4096
#define MTOT (BDIM * TDIM)   // 4096 token rows

__device__ __forceinline__ void async_ld_b128(const void* g, const void* l) {
  asm volatile("global_load_async_to_lds_b128 %0, %1, off"
               :: "v"((unsigned)(uintptr_t)l),
                  "v"((unsigned long long)(uintptr_t)g)
               : "memory");
}
__device__ __forceinline__ void wait_async0() {
  asm volatile("s_wait_asynccnt 0x0" ::: "memory");
}

// ---------------------------------------------------------------------------
// fp32 -> f16 elementwise convert
// ---------------------------------------------------------------------------
__global__ void cvt_f32_f16_kernel(const float* __restrict__ in,
                                   _Float16* __restrict__ out, int n) {
  int i = blockIdx.x * blockDim.x + threadIdx.x;
  if (i < n) out[i] = (_Float16)in[i];
}

// ---------------------------------------------------------------------------
// WMMA GEMM: C[M,N] = A[M,K] @ Bt[N,K]^T  (+bias, optional relu)
// Block tile 128x128, K-step 32, 256 threads = 8 waves in a 2(M) x 4(N) grid,
// each wave computes 64x32 = 4x2 WMMA tiles of 16x16.
// Double-buffered LDS staged through GLOBAL_LOAD_ASYNC_TO_LDS_B128.
// headmode=1: scatter f16 output from [m, h*64+dh] into [B,H,T,DH].
// ---------------------------------------------------------------------------
#define GBM 128
#define GBN 128
#define GBK 32
#define GLD 40   // LDS row stride in halves (padded; 80B, 16B-aligned rows)

__global__ __launch_bounds__(256)
void gemm_xwt_kernel(const _Float16* __restrict__ A,
                     const _Float16* __restrict__ Bt,
                     const float* __restrict__ bias,
                     float* __restrict__ Cf,
                     _Float16* __restrict__ Ch,
                     int M, int N, int K, int relu, int headmode) {
  __shared__ _Float16 As[2][GBM * GLD];
  __shared__ _Float16 Bs[2][GBN * GLD];

  const int tid   = threadIdx.x;
  const int lane  = tid & 31;
  const int wave  = tid >> 5;
  const int wm    = wave >> 2;     // 0..1
  const int wn    = wave & 3;      // 0..3
  const int lmod  = lane & 15;
  const int lhalf = lane >> 4;     // 0 or 1
  const int bm0   = blockIdx.y * GBM;
  const int bn0   = blockIdx.x * GBN;

  // per-thread staging coordinates: tile = 128 rows x 32 halves = 512 uint4;
  // 256 threads handle 2 chunks each (per matrix).
  int srow[2], scol[2];
#pragma unroll
  for (int it = 0; it < 2; ++it) {
    int idx = tid + it * 256;
    srow[it] = idx >> 2;
    scol[it] = (idx & 3) << 3;
  }

  auto issue_stage = [&](int buf, int k0) {
#pragma unroll
    for (int it = 0; it < 2; ++it) {
      const _Float16* ga = A + (size_t)(bm0 + srow[it]) * K + k0 + scol[it];
      async_ld_b128(ga, As[buf] + srow[it] * GLD + scol[it]);
      const _Float16* gb = Bt + (size_t)(bn0 + srow[it]) * K + k0 + scol[it];
      async_ld_b128(gb, Bs[buf] + srow[it] * GLD + scol[it]);
    }
  };

  v8f acc[4][2] = {};
  const int nk = K / GBK;
  issue_stage(0, 0);

  for (int i = 0; i < nk; ++i) {
    const int buf = i & 1;
    wait_async0();       // this wave's copies for tile i are in LDS
    __syncthreads();     // publish; also: everyone finished reading buf^1
    if (i + 1 < nk) issue_stage(buf ^ 1, (i + 1) * GBK);  // overlap with compute

    v16h afr[4], bfr[2];
#pragma unroll
    for (int mt = 0; mt < 4; ++mt) {
      int m = wm * 64 + mt * 16 + lmod;
      const _Float16* p = As[buf] + m * GLD + 8 * lhalf;
      union { v16h v; uint4 u[2]; } fa;
      fa.u[0] = *(const uint4*)(p);
      fa.u[1] = *(const uint4*)(p + 16);
      afr[mt] = fa.v;
    }
#pragma unroll
    for (int nt = 0; nt < 2; ++nt) {
      int n = wn * 32 + nt * 16 + lmod;
      const _Float16* p = Bs[buf] + n * GLD + 8 * lhalf;
      union { v16h v; uint4 u[2]; } fb;
      fb.u[0] = *(const uint4*)(p);
      fb.u[1] = *(const uint4*)(p + 16);
      bfr[nt] = fb.v;
    }
#pragma unroll
    for (int mt = 0; mt < 4; ++mt)
#pragma unroll
      for (int nt = 0; nt < 2; ++nt)
        acc[mt][nt] = __builtin_amdgcn_wmma_f32_16x16x32_f16(
            false, afr[mt], false, bfr[nt], (short)0, acc[mt][nt], false, false);
  }

  // epilogue
#pragma unroll
  for (int mt = 0; mt < 4; ++mt) {
#pragma unroll
    for (int nt = 0; nt < 2; ++nt) {
      int col = bn0 + wn * 32 + nt * 16 + lmod;
      float bv = bias ? bias[col] : 0.0f;
#pragma unroll
      for (int r = 0; r < 8; ++r) {
        int row = bm0 + wm * 64 + mt * 16 + 8 * lhalf + r;
        float v = acc[mt][nt][r] + bv;
        if (relu) v = fmaxf(v, 0.0f);
        if (headmode) {
          int b = row >> 11, t = row & (TDIM - 1);
          int h = col >> 6, dh = col & (DHD - 1);
          Ch[(((size_t)(b * HDIM + h) * TDIM) + t) * DHD + dh] = (_Float16)v;
        } else {
          size_t o = (size_t)row * N + col;
          if (Cf) Cf[o] = v;
          if (Ch) Ch[o] = (_Float16)v;
        }
      }
    }
  }
}

// ---------------------------------------------------------------------------
// Flash-style attention. Grid: (T/64, B*H), 128 threads = 4 waves, each wave
// owns a 16-row q tile; s iterated in tiles of 32. K tile async-staged to LDS;
// V staged transposed through VGPRs. Scores/PV via WMMA; online softmax in
// fp32 with reference masking semantics (masked probs forced to zero).
// ---------------------------------------------------------------------------
#define KS_LD 72   // K tile LDS stride in halves (144B rows, 16B aligned)
#define VT_LD 40   // V^T tile stride
#define PS_LD 40   // per-wave P scratch stride

__global__ __launch_bounds__(128)
void attn_kernel(const _Float16* __restrict__ Qh,
                 const _Float16* __restrict__ Kh,
                 const _Float16* __restrict__ Vh,
                 const unsigned char* __restrict__ mask,
                 _Float16* __restrict__ ctx) {
  __shared__ _Float16 Ks[32 * KS_LD];
  __shared__ _Float16 Vt[64 * VT_LD];
  __shared__ _Float16 Ps[4 * 16 * PS_LD];

  const int bh   = blockIdx.y;           // 0..B*H-1
  const int b    = bh >> 4;
  const int h    = bh & (HDIM - 1);
  const int q0   = blockIdx.x * 64;
  const int tid  = threadIdx.x;
  const int lane = tid & 31;
  const int wave = tid >> 5;
  const int lmod  = lane & 15;
  const int lhalf = lane >> 4;
  const int qbase = q0 + wave * 16;

  const _Float16* Qp = Qh + (size_t)bh * TDIM * DHD;
  const _Float16* Kp = Kh + (size_t)bh * TDIM * DHD;
  const _Float16* Vp = Vh + (size_t)bh * TDIM * DHD;

  // Q fragments (row = qbase + lane%16), pre-scaled by 1/sqrt(DH) = 0.125
  v16h qfr[2];
  {
    int qrow = qbase + lmod;
#pragma unroll
    for (int kc = 0; kc < 2; ++kc) {
      union { v16h v; uint4 u[2]; } f;
      const _Float16* p = Qp + (size_t)qrow * DHD + kc * 32 + 8 * lhalf;
      f.u[0] = *(const uint4*)(p);
      f.u[1] = *(const uint4*)(p + 16);
#pragma unroll
      for (int e = 0; e < 16; ++e) f.v[e] = f.v[e] * (_Float16)0.125f;
      qfr[kc] = f.v;
    }
  }

  float run_max[8], run_sum[8];
#pragma unroll
  for (int r = 0; r < 8; ++r) { run_max[r] = -1e30f; run_sum[r] = 0.0f; }
  v8f cacc[4] = {};

  _Float16* Pw = Ps + wave * 16 * PS_LD;

  for (int s0 = 0; s0 < TDIM; s0 += 32) {
    __syncthreads();  // all waves done reading previous tiles
    // stage K tile [32 s][64 dh] (async) and V^T tile [64 dh][32 s]
#pragma unroll
    for (int it = 0; it < 2; ++it) {
      int idx = tid + it * 128;          // 0..255
      int r = idx >> 3;                  // s row 0..31
      int c = (idx & 7) << 3;            // dh chunk base
      async_ld_b128(Kp + (size_t)(s0 + r) * DHD + c, Ks + r * KS_LD + c);
      uint4 dv = *(const uint4*)(Vp + (size_t)(s0 + r) * DHD + c);
      const _Float16* ph = (const _Float16*)&dv;
#pragma unroll
      for (int j = 0; j < 8; ++j) Vt[(c + j) * VT_LD + r] = ph[j];
    }
    if (s0 + 32 < TDIM) {  // prefetch next tiles into cache
      __builtin_prefetch(Kp + (size_t)(s0 + 32) * DHD + tid * 32);
      __builtin_prefetch(Vp + (size_t)(s0 + 32) * DHD + tid * 32);
    }
    wait_async0();
    __syncthreads();

    // scores S[16 q, 32 s]: two n-tiles, each accumulating over dh in 2 steps
    v8f sfr[2];
#pragma unroll
    for (int nt = 0; nt < 2; ++nt) {
      v8f s = {};
#pragma unroll
      for (int kc = 0; kc < 2; ++kc) {
        union { v16h v; uint4 u[2]; } fb;
        const _Float16* p = Ks + (nt * 16 + lmod) * KS_LD + kc * 32 + 8 * lhalf;
        fb.u[0] = *(const uint4*)(p);
        fb.u[1] = *(const uint4*)(p + 16);
        s = __builtin_amdgcn_wmma_f32_16x16x32_f16(
            false, qfr[kc], false, fb.v, (short)0, s, false, false);
      }
      sfr[nt] = s;
    }

    // online softmax with mask
    float pmat[2][8];
#pragma unroll
    for (int r = 0; r < 8; ++r) {
      int q = qbase + 8 * lhalf + r;
      const unsigned char* mrow = mask + (size_t)q * TDIM + s0;
      float v0 = sfr[0][r];
      float v1 = sfr[1][r];
      v0 = mrow[lmod]      ? -1e30f : v0;
      v1 = mrow[16 + lmod] ? -1e30f : v1;
      float mx = fmaxf(v0, v1);
#pragma unroll
      for (int off = 1; off < 16; off <<= 1)
        mx = fmaxf(mx, __shfl_xor(mx, off, 32));
      float nm = fmaxf(run_max[r], mx);
      float fac = __expf(run_max[r] - nm);
      run_max[r] = nm;
      float p0 = (v0 <= -1e29f) ? 0.0f : __expf(v0 - nm);
      float p1 = (v1 <= -1e29f) ? 0.0f : __expf(v1 - nm);
      pmat[0][r] = p0;
      pmat[1][r] = p1;
      float ps = p0 + p1;
#pragma unroll
      for (int off = 1; off < 16; off <<= 1)
        ps += __shfl_xor(ps, off, 32);
      run_sum[r] = run_sum[r] * fac + ps;
#pragma unroll
      for (int nt = 0; nt < 4; ++nt) cacc[nt][r] *= fac;
    }

    // C-layout -> A-layout relayout of P through per-wave LDS scratch
#pragma unroll
    for (int nt = 0; nt < 2; ++nt)
#pragma unroll
      for (int r = 0; r < 8; ++r)
        Pw[(8 * lhalf + r) * PS_LD + nt * 16 + lmod] = (_Float16)pmat[nt][r];
    asm volatile("s_wait_dscnt 0x0" ::: "memory");
    v16h pfr;
    {
      union { v16h v; uint4 u[2]; } f;
      const _Float16* p = Pw + lmod * PS_LD + 8 * lhalf;
      f.u[0] = *(const uint4*)(p);
      f.u[1] = *(const uint4*)(p + 16);
      pfr = f.v;
    }

    // context += P @ V  (4 dh tiles of 16)
#pragma unroll
    for (int nt = 0; nt < 4; ++nt) {
      union { v16h v; uint4 u[2]; } fb;
      const _Float16* p = Vt + (nt * 16 + lmod) * VT_LD + 8 * lhalf;
      fb.u[0] = *(const uint4*)(p);
      fb.u[1] = *(const uint4*)(p + 16);
      cacc[nt] = __builtin_amdgcn_wmma_f32_16x16x32_f16(
          false, pfr, false, fb.v, (short)0, cacc[nt], false, false);
    }
  }

  // normalize and write context back in [B,T,D] layout (f16 for O-projection)
#pragma unroll
  for (int nt = 0; nt < 4; ++nt) {
#pragma unroll
    for (int r = 0; r < 8; ++r) {
      float denom = run_sum[r];
      float v = (denom > 0.0f) ? cacc[nt][r] / denom : 0.0f;
      int q = qbase + 8 * lhalf + r;
      int col = h * DHD + nt * 16 + lmod;
      ctx[((size_t)b * TDIM + q) * DDIM + col] = (_Float16)v;
    }
  }
}

// ---------------------------------------------------------------------------
// Fused residual + LayerNorm: out = LN(base + alpha*delta) * gamma + beta.
// One block (256 threads) per 1024-element row.
// ---------------------------------------------------------------------------
__global__ __launch_bounds__(256)
void add_ln_kernel(const float* __restrict__ base,
                   const float* __restrict__ delta,
                   const float* __restrict__ alpha_p,
                   const float* __restrict__ gamma,
                   const float* __restrict__ beta,
                   float* __restrict__ outf,
                   _Float16* __restrict__ outh) {
  __shared__ float warp_s[8], warp_s2[8];
  const int row  = blockIdx.x;
  const int tid  = threadIdx.x;
  const int lane = tid & 31;
  const int wave = tid >> 5;
  const float alpha = alpha_p[0];
  const float* bp = base  + (size_t)row * DDIM;
  const float* dp = delta + (size_t)row * DDIM;

  float vals[4];
  float s = 0.0f, s2 = 0.0f;
#pragma unroll
  for (int i = 0; i < 4; ++i) {
    int c = tid + i * 256;
    float v = bp[c] + alpha * dp[c];
    vals[i] = v;
    s += v;
    s2 += v * v;
  }
#pragma unroll
  for (int off = 1; off < 32; off <<= 1) {
    s  += __shfl_xor(s, off, 32);
    s2 += __shfl_xor(s2, off, 32);
  }
  if (lane == 0) { warp_s[wave] = s; warp_s2[wave] = s2; }
  __syncthreads();
  float ts = 0.0f, ts2 = 0.0f;
#pragma unroll
  for (int w = 0; w < 8; ++w) { ts += warp_s[w]; ts2 += warp_s2[w]; }
  const float mu  = ts * (1.0f / DDIM);
  const float var = ts2 * (1.0f / DDIM) - mu * mu;
  const float inv = rsqrtf(var + 1e-5f);
#pragma unroll
  for (int i = 0; i < 4; ++i) {
    int c = tid + i * 256;
    float o = (vals[i] - mu) * inv * gamma[c] + beta[c];
    if (outf) outf[(size_t)row * DDIM + c] = o;
    if (outh) outh[(size_t)row * DDIM + c] = (_Float16)o;
  }
}

// ---------------------------------------------------------------------------
// Host-side orchestration
// ---------------------------------------------------------------------------
extern "C" void kernel_launch(void* const* d_in, const int* in_sizes, int n_in,
                              void* d_out, int out_size, void* d_ws, size_t ws_size,
                              hipStream_t stream) {
  const float* src  = (const float*)d_in[0];
  const unsigned char* mask = (const unsigned char*)d_in[1];
  const float* q_w  = (const float*)d_in[2];
  const float* q_b  = (const float*)d_in[3];
  const float* k_w  = (const float*)d_in[4];
  const float* k_b  = (const float*)d_in[5];
  const float* v_w  = (const float*)d_in[6];
  const float* v_b  = (const float*)d_in[7];
  const float* o_w  = (const float*)d_in[8];
  const float* o_b  = (const float*)d_in[9];
  const float* l1_w = (const float*)d_in[10];
  const float* l1_b = (const float*)d_in[11];
  const float* l2_w = (const float*)d_in[12];
  const float* l2_b = (const float*)d_in[13];
  const float* n1_s = (const float*)d_in[14];
  const float* n1_b = (const float*)d_in[15];
  const float* n2_s = (const float*)d_in[16];
  const float* n2_b = (const float*)d_in[17];
  const float* alpha_attn = (const float*)d_in[18];
  const float* alpha_ff   = (const float*)d_in[19];

  char* wp = (char*)d_ws;
  auto carve = [&](size_t bytes) {
    void* p = (void*)wp;
    wp += (bytes + 255) & ~(size_t)255;
    return p;
  };
  _Float16* Xh   = (_Float16*)carve((size_t)MTOT * DDIM * 2);
  _Float16* Wqh  = (_Float16*)carve((size_t)DDIM * DDIM * 2);
  _Float16* Wkh  = (_Float16*)carve((size_t)DDIM * DDIM * 2);
  _Float16* Wvh  = (_Float16*)carve((size_t)DDIM * DDIM * 2);
  _Float16* Woh  = (_Float16*)carve((size_t)DDIM * DDIM * 2);
  _Float16* W1h  = (_Float16*)carve((size_t)FDIM * DDIM * 2);
  _Float16* W2h  = (_Float16*)carve((size_t)DDIM * FDIM * 2);
  _Float16* Qh   = (_Float16*)carve((size_t)MTOT * DDIM * 2);
  _Float16* Kh   = (_Float16*)carve((size_t)MTOT * DDIM * 2);
  _Float16* Vh   = (_Float16*)carve((size_t)MTOT * DDIM * 2);
  _Float16* ctxh = (_Float16*)carve((size_t)MTOT * DDIM * 2);
  float*    ybuf = (float*)   carve((size_t)MTOT * DDIM * 4);
  float*    x1f  = (float*)   carve((size_t)MTOT * DDIM * 4);
  _Float16* x1h  = (_Float16*)carve((size_t)MTOT * DDIM * 2);
  _Float16* ffh  = (_Float16*)carve((size_t)MTOT * FDIM * 2);

  auto cvt = [&](const float* in, _Float16* out, int n) {
    cvt_f32_f16_kernel<<<(n + 255) / 256, 256, 0, stream>>>(in, out, n);
  };
  cvt(src,  Xh,  MTOT * DDIM);
  cvt(q_w,  Wqh, DDIM * DDIM);
  cvt(k_w,  Wkh, DDIM * DDIM);
  cvt(v_w,  Wvh, DDIM * DDIM);
  cvt(o_w,  Woh, DDIM * DDIM);
  cvt(l1_w, W1h, FDIM * DDIM);
  cvt(l2_w, W2h, DDIM * FDIM);

  // Q/K/V projections: [4096,1024] @ [1024,1024]^T, head-scattered f16 out
  dim3 gproj(DDIM / GBN, MTOT / GBM);
  gemm_xwt_kernel<<<gproj, 256, 0, stream>>>(Xh, Wqh, q_b, nullptr, Qh,
                                             MTOT, DDIM, DDIM, 0, 1);
  gemm_xwt_kernel<<<gproj, 256, 0, stream>>>(Xh, Wkh, k_b, nullptr, Kh,
                                             MTOT, DDIM, DDIM, 0, 1);
  gemm_xwt_kernel<<<gproj, 256, 0, stream>>>(Xh, Wvh, v_b, nullptr, Vh,
                                             MTOT, DDIM, DDIM, 0, 1);

  // attention
  dim3 gattn(TDIM / 64, BDIM * HDIM);
  attn_kernel<<<gattn, 128, 0, stream>>>(Qh, Kh, Vh, mask, ctxh);

  // output projection -> f32
  gemm_xwt_kernel<<<gproj, 256, 0, stream>>>(ctxh, Woh, o_b, ybuf, nullptr,
                                             MTOT, DDIM, DDIM, 0, 0);

  // x1 = LN(src + alpha_attn * attn)
  add_ln_kernel<<<MTOT, 256, 0, stream>>>(src, ybuf, alpha_attn, n1_s, n1_b,
                                          x1f, x1h);

  // FFN1: relu(x1 @ l1_w^T + b1) -> f16 [4096,4096]
  dim3 gff1(FDIM / GBN, MTOT / GBM);
  gemm_xwt_kernel<<<gff1, 256, 0, stream>>>(x1h, W1h, l1_b, nullptr, ffh,
                                            MTOT, FDIM, DDIM, 1, 0);
  // FFN2: ff @ l2_w^T + b2 -> f32 [4096,1024]
  gemm_xwt_kernel<<<gproj, 256, 0, stream>>>(ffh, W2h, l2_b, ybuf, nullptr,
                                             MTOT, DDIM, FDIM, 0, 0);

  // out = LN(x1 + alpha_ff * ff)
  add_ln_kernel<<<MTOT, 256, 0, stream>>>(x1f, ybuf, alpha_ff, n2_s, n2_b,
                                          (float*)d_out, nullptr);
}
